// PersonalizedAttention_49014166782505
// MI455X (gfx1250) — matrix operand order repaired and checked by reading, not compiled, not run on hardware
//
#include <hip/hip_runtime.h>
#include <hip/hip_bf16.h>
#include <math.h>

// B=4096, S=50, D=768, U=768, Q=200
#define BB 4096
#define SS 50
#define DD 768
#define UU 768
#define QQ 200
#define QPAD 224            // Q padded to a multiple of 32 for the WMMA K-loop
#define KS1 (UU / 32)       // 24 k-steps for GEMM1
#define KS2 (QPAD / 32)     // 7  k-steps for GEMM2
#define NT1P 16             // n-tiles GEMM1, padded 13->16 (4 per wave, no guard)
#define NT2 (DD / 16)       // 48 n-tiles, GEMM2 (6 per wave, exact)
#define N1FRAG (NT1P * KS1 * 32 * 16)  // 196608 f16 elements of W1frag
#define N2FRAG (NT2 * KS2 * 32 * 16)   // 172032 f16 elements of W2frag

typedef __attribute__((ext_vector_type(16))) _Float16 v16h;
typedef __attribute__((ext_vector_type(8)))  float    v8f;

// Fragment-order index helpers (wave32; contiguous k order for A and B, as used here):
//  A 16x32 f16 : lane = 16h + m, element i -> K = 16h + i (contiguous run)
//  B 32x16 f16 : lane = 16h + n, element i -> K = 16h + i (contiguous run)
//  C/D 16x16 f32: lane = 16h + n, VGPR v -> M = v + 8h

// ---------------------------------------------------------------------------
// Prepass: convert W1 [Q,U] and W2 [D,Q] fp32 -> f16 in exact B-fragment
// order, with all boundary zero-padding resolved here (hot loops branch-free).
// ---------------------------------------------------------------------------
__global__ __launch_bounds__(256) void pa_wfrag(const float* __restrict__ W1,
                                                const float* __restrict__ W2,
                                                _Float16* __restrict__ W1f,
                                                _Float16* __restrict__ W2f) {
  int e = blockIdx.x * 256 + threadIdx.x;
  if (e < N1FRAG) {
    const int i = e & 15, lane = (e >> 4) & 31, t = e >> 9;
    const int ks = t % KS1, nt = t / KS1;
    const int h = lane >> 4, mr = lane & 15;
    const int n = nt * 16 + mr;
    const int k = ks * 32 + 16 * h + i;           // < 768 always
    W1f[e] = (n < QQ) ? (_Float16)W1[(size_t)n * UU + k] : (_Float16)0.f;
    return;
  }
  e -= N1FRAG;
  if (e < N2FRAG) {
    const int i = e & 15, lane = (e >> 4) & 31, t = e >> 9;
    const int ks = t % KS2, nt = t / KS2;
    const int h = lane >> 4, mr = lane & 15;
    const int n = nt * 16 + mr;                   // < 768 always
    const int k = ks * 32 + 16 * h + i;
    W2f[e] = (k < QQ) ? (_Float16)W2[(size_t)n * QQ + k] : (_Float16)0.f;
  }
}

// inverse A-layout map: tile element (m, k) -> fragment slot (ks, lane, i)
__device__ __forceinline__ int a_frag_index(int m, int k) {
  const int ks = k >> 5, ko = k & 31;
  const int h = ko >> 4;                          // 0..15 -> h0 ; 16..31 -> h1
  const int i = ko & 15;
  const int lane = h * 16 + m;
  return ((ks * 32 + lane) << 4) + i;
}

// ---------------------------------------------------------------------------
// Kernel 1: q = relu(ue @ W1^T + b1) -> [B, QPAD] f16 (pad cols zeroed).
// One block = 16 rows of B; 4 waves; wave w owns n-tiles {4w .. 4w+3}.
// Hot loop is branch-free: tiles beyond Q multiply zero-padded B fragments.
// ---------------------------------------------------------------------------
__global__ __launch_bounds__(128) void pa_mlp1(const float* __restrict__ ue,
                                               const _Float16* __restrict__ W1f,
                                               const float* __restrict__ b1,
                                               _Float16* __restrict__ qout) {
  __shared__ __align__(32) _Float16 ueFrag[KS1 * 32 * 16];  // 24.5 KB
  const int tid = threadIdx.x;
  const int mBase = blockIdx.x * 16;

  // coalesced global read, permuted LDS scatter into A-fragment order
  for (int idx = tid; idx < 16 * UU; idx += 128) {
    const int m = idx / UU, k = idx - m * UU;
    ueFrag[a_frag_index(m, k)] = (_Float16)ue[(size_t)mBase * UU + idx];
  }
  __syncthreads();

  const int lane = tid & 31, wave = tid >> 5;
  const int h = lane >> 4, mr = lane & 15;
  const int ntBase = wave * 4;

  v8f acc[4] = {{}, {}, {}, {}};
  for (int ks = 0; ks < KS1; ++ks) {
    const v16h a = *(const v16h*)&ueFrag[(ks * 32 + lane) << 4];
#pragma unroll
    for (int j = 0; j < 4; ++j) {
      const v16h b = *(const v16h*)
          &W1f[(((size_t)(ntBase + j) * KS1 + ks) * 32 + lane) << 4];
      acc[j] = __builtin_amdgcn_wmma_f32_16x16x32_f16(
          false, a, false, b, (short)0, acc[j], false, false);
    }
  }
#pragma unroll
  for (int j = 0; j < 4; ++j) {
    const int n = (ntBase + j) * 16 + mr;
    if (n < QPAD) {                               // tile 13 stores the zero pad
      const bool nv = (n < QQ);
      const float bias = nv ? b1[n] : 0.f;
#pragma unroll
      for (int v = 0; v < 8; ++v) {               // D: M = v + 8h, N = mr
        const int m = mBase + v + 8 * h;
        const float val = nv ? fmaxf(acc[j][v] + bias, 0.f) : 0.f;
        qout[(size_t)m * QPAD + n] = (_Float16)val;
      }
    }
  }
}

// ---------------------------------------------------------------------------
// Kernel 2: w = tanh(q @ W2^T + b2) -> [B, D] fp32.
// One block = 16 rows of B; 8 waves; wave w owns n-tiles {6w .. 6w+5}.
// ---------------------------------------------------------------------------
__global__ __launch_bounds__(256) void pa_mlp2(const _Float16* __restrict__ qin,
                                               const _Float16* __restrict__ W2f,
                                               const float* __restrict__ b2,
                                               float* __restrict__ wout) {
  __shared__ __align__(32) _Float16 qFrag[KS2 * 32 * 16];   // 7 KB
  const int tid = threadIdx.x;
  const int mBase = blockIdx.x * 16;

  for (int idx = tid; idx < 16 * QPAD; idx += 256) {
    const int m = idx / QPAD, k = idx - m * QPAD;
    qFrag[a_frag_index(m, k)] = qin[(size_t)mBase * QPAD + idx];
  }
  __syncthreads();

  const int lane = tid & 31, wave = tid >> 5;
  const int h = lane >> 4, mr = lane & 15;
  const int ntBase = wave * 6;

  v8f acc[6] = {{}, {}, {}, {}, {}, {}};
  for (int ks = 0; ks < KS2; ++ks) {
    const v16h a = *(const v16h*)&qFrag[(ks * 32 + lane) << 4];
#pragma unroll
    for (int j = 0; j < 6; ++j) {
      const v16h b = *(const v16h*)
          &W2f[(((size_t)(ntBase + j) * KS2 + ks) * 32 + lane) << 4];
      acc[j] = __builtin_amdgcn_wmma_f32_16x16x32_f16(
          false, a, false, b, (short)0, acc[j], false, false);
    }
  }
#pragma unroll
  for (int j = 0; j < 6; ++j) {
    const int n = (ntBase + j) * 16 + mr;         // < 768 always
    const float bias = b2[n];
#pragma unroll
    for (int v = 0; v < 8; ++v) {
      const int m = mBase + v + 8 * h;
      wout[(size_t)m * DD + n] = tanhf(acc[j][v] + bias);
    }
  }
}

// ---------------------------------------------------------------------------
// Kernel 3 (the memory-bound part): per batch row b, stage c[b] (153.6 KB)
// in LDS once (2 WGs fit in the 320 KB WGP LDS) using the CDNA5 async
// global->LDS path (no VGPR bounce, deep ASYNCcnt queue), compute the 50
// scores vs w[b], softmax over S inside one wave32, rescale from LDS, stream
// out.  c touches HBM exactly once each way -> 1.26 GB floor ~= 54 us.
// NOTE: no static __shared__ in this kernel, so the dynamic-LDS base is at
// LDS byte offset 0 and plain byte offsets can be used as async LDS dests.
// ---------------------------------------------------------------------------
__global__ __launch_bounds__(256) void pa_attn(const float* __restrict__ c,
                                               const float* __restrict__ wvec,
                                               float* __restrict__ out) {
  extern __shared__ float smem[];
  float* cT   = smem;                 // 50*768 floats = 153600 B (LDS off 0)
  float* wsh  = smem + SS * DD;       // 768 floats
  float* attn = wsh + DD;             // 64 floats (50 used)

  const int tid = threadIdx.x;
  const int b = blockIdx.x;
  const float* cb = c + (size_t)b * SS * DD;

  for (int i = tid; i < DD; i += 256) wsh[i] = wvec[(size_t)b * DD + i];

  // async-stage c[b] into LDS: 16 B per lane per issue, tracked by ASYNCcnt
  const int n4 = SS * DD / 4;         // 9600 16-byte chunks
  {
    const unsigned long long gbase = (unsigned long long)cb;
    for (int i = tid; i < n4; i += 256) {
      const unsigned int lds_off = (unsigned int)i * 16u;
      const unsigned long long gaddr = gbase + (unsigned long long)i * 16u;
      asm volatile("global_load_async_to_lds_b128 %0, %1, off"
                   :: "v"(lds_off), "v"(gaddr) : "memory");
    }
    asm volatile("s_wait_asynccnt 0x0" ::: "memory");
  }
  __syncthreads();

  const int lane = tid & 31, wave = tid >> 5;

  for (int s = wave; s < SS; s += 8) {
    const float* row = cT + s * DD;
    float acc = 0.f;
    for (int d = lane; d < DD; d += 32) acc += row[d] * wsh[d];
#pragma unroll
    for (int off = 16; off > 0; off >>= 1) acc += __shfl_xor(acc, off, 32);
    if (lane == 0) attn[s] = acc;
  }
  __syncthreads();

  if (wave == 0) {                    // softmax over S=50 in one wave32
    float v0 = (lane < SS) ? attn[lane] : -1e30f;
    float v1 = (lane + 32 < SS) ? attn[lane + 32] : -1e30f;
    float mx = fmaxf(v0, v1);
#pragma unroll
    for (int off = 16; off > 0; off >>= 1) mx = fmaxf(mx, __shfl_xor(mx, off, 32));
    float e0 = (lane < SS) ? __expf(v0 - mx) : 0.f;
    float e1 = (lane + 32 < SS) ? __expf(v1 - mx) : 0.f;
    float sum = e0 + e1;
#pragma unroll
    for (int off = 16; off > 0; off >>= 1) sum += __shfl_xor(sum, off, 32);
    const float inv = 1.f / sum;
    if (lane < SS)      attn[lane]      = e0 * inv;
    if (lane + 32 < SS) attn[lane + 32] = e1 * inv;
  }
  __syncthreads();

  const float4* cT4 = (const float4*)cT;
  float4* out4 = (float4*)(out + (size_t)b * SS * DD);
  for (int i = tid; i < n4; i += 256) {
    const float a = attn[i / (DD / 4)];
    float4 v = cT4[i];
    v.x *= a; v.y *= a; v.z *= a; v.w *= a;
    out4[i] = v;
  }
}

// ---------------------------------------------------------------------------
extern "C" void kernel_launch(void* const* d_in, const int* in_sizes, int n_in,
                              void* d_out, int out_size, void* d_ws, size_t ws_size,
                              hipStream_t stream) {
  (void)in_sizes; (void)n_in; (void)out_size; (void)ws_size;
  const float* c  = (const float*)d_in[0];   // [B,S,D]
  const float* ue = (const float*)d_in[1];   // [B,U]
  const float* W1 = (const float*)d_in[2];   // [Q,U]
  const float* b1 = (const float*)d_in[3];   // [Q]
  const float* W2 = (const float*)d_in[4];   // [D,Q]
  const float* b2 = (const float*)d_in[5];   // [D]
  float* out = (float*)d_out;                // [B,S,D]

  // d_ws layout (all 32B aligned):
  char* ws = (char*)d_ws;
  _Float16* qpad = (_Float16*)ws;                          // B*QPAD f16
  ws += (size_t)BB * QPAD * sizeof(_Float16);              // 1,835,008
  float* wv = (float*)ws;                                  // B*D f32
  ws += (size_t)BB * DD * sizeof(float);                   // 12,582,912
  _Float16* W1f = (_Float16*)ws;                           // 393,216 B
  ws += (size_t)N1FRAG * sizeof(_Float16);
  _Float16* W2f = (_Float16*)ws;                           // 344,064 B

  const int nfrag = N1FRAG + N2FRAG;
  pa_wfrag<<<(nfrag + 255) / 256, 256, 0, stream>>>(W1, W2, W1f, W2f);
  pa_mlp1<<<BB / 16, 128, 0, stream>>>(ue, W1f, b1, qpad);
  pa_mlp2<<<BB / 16, 256, 0, stream>>>(qpad, W2f, b2, wv);

  const size_t sh = (size_t)(SS * DD + DD + 64) * sizeof(float);  // ~157 KB
  pa_attn<<<BB, 256, sh, stream>>>(c, wv, out);
}
